// Attention_74766790689101
// MI455X (gfx1250) — matrix operand order repaired and checked by reading, not compiled, not run on hardware
//
#include <hip/hip_runtime.h>

typedef _Float16 v16h __attribute__((ext_vector_type(16)));
typedef _Float16 h8   __attribute__((ext_vector_type(8)));
typedef float    v8f  __attribute__((ext_vector_type(8)));

#define DIM   768
#define NHEAD 12
#define HDIM  64
#define BATCH 4
#define SEQ   2048
#define TOK   (BATCH*SEQ)    /* 8192 */
#define QKVN  (3*DIM)        /* 2304 */
#define ATTN_SCALE 0.125f    /* 64^-0.5 */

#define WMMA_F16(a,b,c) __builtin_amdgcn_wmma_f32_16x16x32_f16(false,(a),false,(b),(short)0,(c),false,false)

// ---------------- convert f32 -> f16, same layout ----------------
__global__ __launch_bounds__(256) void cvt_f16(const float* __restrict__ src,
                                               _Float16* __restrict__ dst, int n) {
  for (int i = blockIdx.x * blockDim.x + threadIdx.x; i < n; i += gridDim.x * blockDim.x)
    dst[i] = (_Float16)src[i];
}

// ------------- transpose-convert: dst[n*K+k] = (f16)src[k*N+n] -------------
__global__ __launch_bounds__(256) void cvtT_f16(const float* __restrict__ src,
                                                _Float16* __restrict__ dst, int K, int N) {
  int total = K * N;
  for (int i = blockIdx.x * blockDim.x + threadIdx.x; i < total; i += gridDim.x * blockDim.x) {
    int n = i % N, k = i / N;
    dst[(size_t)n * K + k] = (_Float16)src[i];
  }
}

// build 16x32 A-fragment for one 16-row group
__device__ __forceinline__ v16h load_afrag(const _Float16* row, int k0, int half) {
  h8 c0 = *(const h8*)(row + k0 + 16 * half);
  h8 c1 = *(const h8*)(row + k0 + 16 * half + 8);
  v16h a;
#pragma unroll
  for (int i = 0; i < 8; i++) { a[i] = c0[i]; a[8 + i] = c1[i]; }
  return a;
}

// ---------------- QKV GEMM (32x32 tile per wave) ----------------
__device__ __forceinline__ void qkv_store(_Float16* __restrict__ qb, _Float16* __restrict__ kb,
                                          _Float16* __restrict__ vTb,
                                          int c, int mbase, int half, const v8f& acc) {
  const int which = c / DIM;           // 0=q 1=k 2=v
  const int rem   = c % DIM;
  const int h     = rem / HDIM, d = rem % HDIM;
#pragma unroll
  for (int r = 0; r < 8; r++) {
    int m  = mbase + r + 8 * half;
    int bI = m >> 11, n = m & (SEQ - 1);
    int bh = bI * NHEAD + h;
    float v = acc[r];
    if (which == 0)      qb [((size_t)bh * SEQ + n) * HDIM + d] = (_Float16)(v * ATTN_SCALE);
    else if (which == 1) kb [((size_t)bh * SEQ + n) * HDIM + d] = (_Float16)v;
    else                 vTb[((size_t)bh * HDIM + d) * SEQ + n] = (_Float16)v;
  }
}

__global__ __launch_bounds__(256) void qkv_gemm(const _Float16* __restrict__ xh,
                                                const _Float16* __restrict__ wT,   // [QKVN, DIM]
                                                _Float16* __restrict__ qb,
                                                _Float16* __restrict__ kb,
                                                _Float16* __restrict__ vTb) {
  const int lane = threadIdx.x & 31;
  const int wave = threadIdx.x >> 5;
  const int w    = blockIdx.x * 8 + wave;
  const int nt   = QKVN / 32;                 // 72
  const int m0 = (w / nt) * 32;
  const int n0 = (w % nt) * 32;
  const int half = lane >> 4, l16 = lane & 15;

  const _Float16* arow0 = xh + (size_t)(m0 + l16) * DIM;
  const _Float16* arow1 = xh + (size_t)(m0 + 16 + l16) * DIM;
  const _Float16* brow0 = wT + (size_t)(n0 + l16) * DIM;
  const _Float16* brow1 = wT + (size_t)(n0 + 16 + l16) * DIM;

  v8f a00 = {}, a01 = {}, a10 = {}, a11 = {};
  for (int k0 = 0; k0 < DIM; k0 += 32) {
    v16h fa0 = load_afrag(arow0, k0, half);
    v16h fa1 = load_afrag(arow1, k0, half);
    v16h fb0 = *(const v16h*)(brow0 + k0 + 16 * half);
    v16h fb1 = *(const v16h*)(brow1 + k0 + 16 * half);
    a00 = WMMA_F16(fa0, fb0, a00);
    a01 = WMMA_F16(fa0, fb1, a01);
    a10 = WMMA_F16(fa1, fb0, a10);
    a11 = WMMA_F16(fa1, fb1, a11);
  }
  qkv_store(qb, kb, vTb, n0 + l16,      m0,      half, a00);
  qkv_store(qb, kb, vTb, n0 + 16 + l16, m0,      half, a01);
  qkv_store(qb, kb, vTb, n0 + l16,      m0 + 16, half, a10);
  qkv_store(qb, kb, vTb, n0 + 16 + l16, m0 + 16, half, a11);
}

// ---------------- Flash attention with async-to-LDS double buffering ----------------
__global__ __launch_bounds__(256) void flash_attn(const _Float16* __restrict__ qb,
                                                  const _Float16* __restrict__ kb,
                                                  const _Float16* __restrict__ vTb,
                                                  _Float16* __restrict__ ctx) {
  constexpr int KS = 80;  // padded LDS row stride for K tile (f16 elems, 160B)
  constexpr int VS = 48;  // padded LDS row stride for V tile (96B)
  __shared__ __align__(128) _Float16 ksh[2][32 * KS];   // 32 kv rows x 64 d
  __shared__ __align__(128) _Float16 vsh[2][64 * VS];   // 64 d rows x 32 j
  __shared__ __align__(128) _Float16 plds[8][16][32];   // per-wave P transpose staging

  const int tid  = threadIdx.x;
  const int lane = tid & 31, wave = tid >> 5;
  const int half = lane >> 4, l16 = lane & 15;
  const int bh   = blockIdx.x >> 4;
  const int slab = blockIdx.x & 15;
  const int bI   = bh / NHEAD, h = bh % NHEAD;
  const int n0   = slab * 128 + wave * 16;

  const _Float16* kbase = kb  + (size_t)bh * SEQ * HDIM;
  const _Float16* vbase = vTb + (size_t)bh * HDIM * SEQ;

  // cooperative async staging: 256 threads x one b128 each per tile
  const int krow = tid >> 3, kseg = tid & 7;   // K: 32 rows x 8 x 16B
  const int vrow = tid >> 2, vseg = tid & 3;   // V: 64 rows x 4 x 16B

  auto issue_tile = [&](int buf, int j0) {
    const _Float16* gk = kbase + (size_t)(j0 + krow) * HDIM + kseg * 8;
    unsigned lk = (unsigned)(size_t)&ksh[buf][krow * KS + kseg * 8];
    asm volatile("global_load_async_to_lds_b128 %0, %1, off" :: "v"(lk), "v"(gk) : "memory");
    const _Float16* gv = vbase + (size_t)vrow * SEQ + j0 + vseg * 8;
    unsigned lv = (unsigned)(size_t)&vsh[buf][vrow * VS + vseg * 8];
    asm volatile("global_load_async_to_lds_b128 %0, %1, off" :: "v"(lv), "v"(gv) : "memory");
  };

  // Q fragments (A-layout), d-chunks 0..31 and 32..63
  v16h aq0, aq1;
  {
    const _Float16* qrow = qb + ((size_t)bh * SEQ + n0 + l16) * HDIM;
    aq0 = load_afrag(qrow, 0, half);
    aq1 = load_afrag(qrow, 32, half);
  }

  float mrow[8], lrow[8];
  v8f o0 = {}, o1 = {}, o2 = {}, o3 = {};
#pragma unroll
  for (int r = 0; r < 8; r++) { mrow[r] = -1e30f; lrow[r] = 0.0f; }

  issue_tile(0, 0);

  constexpr int NSTEP = SEQ / 32;   // 64
  for (int jt = 0; jt < NSTEP; ++jt) {
    const int buf = jt & 1;
    if (jt + 1 < NSTEP) {
      issue_tile(buf ^ 1, (jt + 1) * 32);       // prefetch next tile (other buffer)
      asm volatile("s_wait_asynccnt 0x2" ::: "memory");  // current tile's 2 DMAs done
    } else {
      asm volatile("s_wait_asynccnt 0x0" ::: "memory");
    }
    __syncthreads();                            // everyone's share landed

    const _Float16* kt = &ksh[buf][0];
    const _Float16* vt = &vsh[buf][0];

    // ---- S = Q K^T for this 32-kv tile ----
    v8f s0 = {}, s1 = {};
    {
      v16h b00 = *(const v16h*)(kt + (size_t)l16 * KS + 16 * half);
      v16h b01 = *(const v16h*)(kt + (size_t)l16 * KS + 32 + 16 * half);
      v16h b10 = *(const v16h*)(kt + (size_t)(16 + l16) * KS + 16 * half);
      v16h b11 = *(const v16h*)(kt + (size_t)(16 + l16) * KS + 32 + 16 * half);
      s0 = WMMA_F16(aq0, b00, s0);
      s0 = WMMA_F16(aq1, b01, s0);
      s1 = WMMA_F16(aq0, b10, s1);
      s1 = WMMA_F16(aq1, b11, s1);
    }

    // ---- online softmax (row reductions across 16 lanes) ----
    float alpha[8];
#pragma unroll
    for (int r = 0; r < 8; r++) {
      float t = fmaxf(s0[r], s1[r]);
      t = fmaxf(t, __shfl_xor(t, 1));
      t = fmaxf(t, __shfl_xor(t, 2));
      t = fmaxf(t, __shfl_xor(t, 4));
      t = fmaxf(t, __shfl_xor(t, 8));
      float mn = fmaxf(mrow[r], t);
      alpha[r] = __expf(mrow[r] - mn);
      mrow[r]  = mn;
      s0[r] = __expf(s0[r] - mn);
      s1[r] = __expf(s1[r] - mn);
      float rs = s0[r] + s1[r];
      rs += __shfl_xor(rs, 1);
      rs += __shfl_xor(rs, 2);
      rs += __shfl_xor(rs, 4);
      rs += __shfl_xor(rs, 8);
      lrow[r] = lrow[r] * alpha[r] + rs;
      plds[wave][r + 8 * half][l16]      = (_Float16)s0[r];
      plds[wave][r + 8 * half][16 + l16] = (_Float16)s1[r];
    }

    // ---- reload P in A-layout (same wave) ----
    v16h ap;
    {
      const _Float16* pr = &plds[wave][l16][0];
      h8 c0 = *(const h8*)(pr + 16 * half);
      h8 c1 = *(const h8*)(pr + 16 * half + 8);
#pragma unroll
      for (int i = 0; i < 8; i++) { ap[i] = c0[i]; ap[8 + i] = c1[i]; }
    }

    // ---- rescale O, then O += P @ V ----
#pragma unroll
    for (int r = 0; r < 8; r++) {
      o0[r] *= alpha[r]; o1[r] *= alpha[r];
      o2[r] *= alpha[r]; o3[r] *= alpha[r];
    }
    v16h bv0 = *(const v16h*)(vt + (size_t)(l16)      * VS + 16 * half);
    v16h bv1 = *(const v16h*)(vt + (size_t)(16 + l16) * VS + 16 * half);
    v16h bv2 = *(const v16h*)(vt + (size_t)(32 + l16) * VS + 16 * half);
    v16h bv3 = *(const v16h*)(vt + (size_t)(48 + l16) * VS + 16 * half);
    o0 = WMMA_F16(ap, bv0, o0);
    o1 = WMMA_F16(ap, bv1, o1);
    o2 = WMMA_F16(ap, bv2, o2);
    o3 = WMMA_F16(ap, bv3, o3);

    __syncthreads();   // all waves done consuming buf before it is refilled
  }

  // ---- epilogue: normalize, write ctx[b, n, h*64 + d] ----
#pragma unroll
  for (int r = 0; r < 8; r++) {
    int n = n0 + r + 8 * half;
    float inv = 1.0f / lrow[r];
    _Float16* cb = ctx + ((size_t)bI * SEQ + n) * DIM + h * HDIM + l16;
    cb[0]  = (_Float16)(o0[r] * inv);
    cb[16] = (_Float16)(o1[r] * inv);
    cb[32] = (_Float16)(o2[r] * inv);
    cb[48] = (_Float16)(o3[r] * inv);
  }
}

// ---------------- Proj GEMM (32x32 tile per wave) + bias, f32 out ----------------
__global__ __launch_bounds__(256) void proj_gemm(const _Float16* __restrict__ ctx,
                                                 const _Float16* __restrict__ wT,   // [DIM, DIM]
                                                 const float* __restrict__ bias,
                                                 float* __restrict__ out) {
  const int lane = threadIdx.x & 31;
  const int wave = threadIdx.x >> 5;
  const int w    = blockIdx.x * 8 + wave;
  const int nt   = DIM / 32;                    // 24
  const int m0 = (w / nt) * 32;
  const int n0 = (w % nt) * 32;
  const int half = lane >> 4, l16 = lane & 15;

  const _Float16* arow0 = ctx + (size_t)(m0 + l16) * DIM;
  const _Float16* arow1 = ctx + (size_t)(m0 + 16 + l16) * DIM;
  const _Float16* brow0 = wT  + (size_t)(n0 + l16) * DIM;
  const _Float16* brow1 = wT  + (size_t)(n0 + 16 + l16) * DIM;

  v8f a00 = {}, a01 = {}, a10 = {}, a11 = {};
  for (int k0 = 0; k0 < DIM; k0 += 32) {
    v16h fa0 = load_afrag(arow0, k0, half);
    v16h fa1 = load_afrag(arow1, k0, half);
    v16h fb0 = *(const v16h*)(brow0 + k0 + 16 * half);
    v16h fb1 = *(const v16h*)(brow1 + k0 + 16 * half);
    a00 = WMMA_F16(fa0, fb0, a00);
    a01 = WMMA_F16(fa0, fb1, a01);
    a10 = WMMA_F16(fa1, fb0, a10);
    a11 = WMMA_F16(fa1, fb1, a11);
  }

  const int c0 = n0 + l16, c1 = n0 + 16 + l16;
  const float bc0 = bias[c0], bc1 = bias[c1];
#pragma unroll
  for (int r = 0; r < 8; r++) {
    int ma = m0 + r + 8 * half;
    int mb = m0 + 16 + r + 8 * half;
    out[(size_t)ma * DIM + c0] = a00[r] + bc0;
    out[(size_t)ma * DIM + c1] = a01[r] + bc1;
    out[(size_t)mb * DIM + c0] = a10[r] + bc0;
    out[(size_t)mb * DIM + c1] = a11[r] + bc1;
  }
}

extern "C" void kernel_launch(void* const* d_in, const int* in_sizes, int n_in,
                              void* d_out, int out_size, void* d_ws, size_t ws_size,
                              hipStream_t stream) {
  const float* x     = (const float*)d_in[0];   // [4,2048,768]
  const float* Wqkv  = (const float*)d_in[1];   // [768,2304]
  const float* Wproj = (const float*)d_in[2];   // [768,768]
  const float* bproj = (const float*)d_in[3];   // [768]
  float* out = (float*)d_out;

  char* ws = (char*)d_ws;
  size_t off = 0;
  _Float16* xh     = (_Float16*)(ws + off); off += (size_t)TOK  * DIM  * 2;
  _Float16* wqkvT  = (_Float16*)(ws + off); off += (size_t)QKVN * DIM  * 2;
  _Float16* wprojT = (_Float16*)(ws + off); off += (size_t)DIM  * DIM  * 2;
  _Float16* qb     = (_Float16*)(ws + off); off += (size_t)BATCH * NHEAD * SEQ * HDIM * 2;
  _Float16* kb     = (_Float16*)(ws + off); off += (size_t)BATCH * NHEAD * SEQ * HDIM * 2;
  _Float16* vTb    = (_Float16*)(ws + off); off += (size_t)BATCH * NHEAD * SEQ * HDIM * 2;
  _Float16* ctxh   = (_Float16*)(ws + off); off += (size_t)TOK  * DIM  * 2;

  cvt_f16 <<<2048, 256, 0, stream>>>(x, xh, TOK * DIM);
  cvtT_f16<<<1024, 256, 0, stream>>>(Wqkv,  wqkvT,  DIM, QKVN);
  cvtT_f16<<< 512, 256, 0, stream>>>(Wproj, wprojT, DIM, DIM);

  // 256 x 72 tiles of 32x32, 8 waves/block
  qkv_gemm<<<(256 * 72) / 8, 256, 0, stream>>>(xh, wqkvT, qb, kb, vTb);

  // 48 (b,h) x 16 slabs of 128 query rows
  flash_attn<<<BATCH * NHEAD * (SEQ / 128), 256, 0, stream>>>(qb, kb, vTb, ctxh);

  // 256 x 24 tiles of 32x32, 8 waves/block
  proj_gemm<<<(256 * 24) / 8, 256, 0, stream>>>(ctxh, wprojT, bproj, out);
}
